// _FakeCceOutputLayer_57483842289900
// MI455X (gfx1250) — matrix operand order repaired and checked by reading, not compiled, not run on hardware
//
#include <hip/hip_runtime.h>
#include <hip/hip_bf16.h>
#include <math.h>

#define SEQ   2048
#define BATCH 2
#define HID   2048
#define VOCAB 50257
#define NTOK  (SEQ * BATCH)
#define IGNORE_INDEX (-100)

#define ROWS  64                      // tokens per block (4 x 16-row WMMA subtiles)
#define NSLAB 8                       // vocab slabs (grid.y)
#define SLAB  ((VOCAB + NSLAB - 1) / NSLAB)   // 6283
#define WAVES 16
#define TPB   (WAVES * 32)            // 512 threads

typedef __attribute__((ext_vector_type(16))) __bf16 v16bf;
typedef __attribute__((ext_vector_type(8)))  __bf16 v8bf;
typedef __attribute__((ext_vector_type(8)))  float  v8f;

union ABfrag { v16bf v; v8bf h[2]; };

// ---------------------------------------------------------------------------
// Kernel 0: W fp32 -> bf16 (one vocab row per block, 8 elems per thread).
// ---------------------------------------------------------------------------
__global__ __launch_bounds__(256) void wcvt_kernel(
    const float* __restrict__ w, __bf16* __restrict__ wbf)
{
    size_t base = (size_t)blockIdx.x * HID + threadIdx.x * 8;
    const float4* src = (const float4*)(w + base);
    float4 a = src[0], b = src[1];
    v8bf o;
    o[0] = (__bf16)a.x; o[1] = (__bf16)a.y; o[2] = (__bf16)a.z; o[3] = (__bf16)a.w;
    o[4] = (__bf16)b.x; o[5] = (__bf16)b.y; o[6] = (__bf16)b.z; o[7] = (__bf16)b.w;
    *(v8bf*)(wbf + base) = o;
}

// B-fragment loaders: bf16 path = two b128 loads; fp32 path = 4x float4 + cvt.
__device__ __forceinline__ void load_bfrag(ABfrag& b, const __bf16* wrow, int koff) {
    b.h[0] = *(const v8bf*)(wrow + koff);
    b.h[1] = *(const v8bf*)(wrow + koff + 8);
}
__device__ __forceinline__ void load_bfrag(ABfrag& b, const float* wrow, int koff) {
    const float4* wp = (const float4*)(wrow + koff);
    float4 w0 = wp[0], w1 = wp[1], w2 = wp[2], w3 = wp[3];
    b.v[0]  = (__bf16)w0.x; b.v[1]  = (__bf16)w0.y;
    b.v[2]  = (__bf16)w0.z; b.v[3]  = (__bf16)w0.w;
    b.v[4]  = (__bf16)w1.x; b.v[5]  = (__bf16)w1.y;
    b.v[6]  = (__bf16)w1.z; b.v[7]  = (__bf16)w1.w;
    b.v[8]  = (__bf16)w2.x; b.v[9]  = (__bf16)w2.y;
    b.v[10] = (__bf16)w2.z; b.v[11] = (__bf16)w2.w;
    b.v[12] = (__bf16)w3.x; b.v[13] = (__bf16)w3.y;
    b.v[14] = (__bf16)w3.z; b.v[15] = (__bf16)w3.w;
}

// ---------------------------------------------------------------------------
// Kernel 1: per (token-tile, vocab-slab) online logsumexp partials via WMMA.
// Templated on weight element type (bf16 workspace copy, or fp32 fallback).
// ---------------------------------------------------------------------------
template <typename WT>
__global__ __launch_bounds__(TPB) void cce_partial_kernel(
    const float* __restrict__ x,      // [NTOK, HID] fp32
    const WT*    __restrict__ w,      // [VOCAB, HID]
    float* __restrict__ pmax,         // [NTOK, NSLAB]
    float* __restrict__ psum)         // [NTOK, NSLAB]
{
    __shared__ __bf16 xt[ROWS * HID];            // 256 KB bf16 x-tile
    __shared__ float  redmax[WAVES][ROWS];
    __shared__ float  redsum[WAVES][ROWS];

    const int tid  = threadIdx.x;
    const int lane = tid & 31;
    const int wave = tid >> 5;
    const int nb   = blockIdx.x * ROWS;          // first token row of this tile

    // Stage x tile into LDS as bf16 (vectorized fp32 reads).
    const float4* xg = (const float4*)(x + (size_t)nb * HID);
    for (int i = tid; i < ROWS * HID / 4; i += TPB) {
        float4 f = xg[i];
        int o = i * 4;
        xt[o + 0] = (__bf16)f.x; xt[o + 1] = (__bf16)f.y;
        xt[o + 2] = (__bf16)f.z; xt[o + 3] = (__bf16)f.w;
    }
    __syncthreads();

    const int rsub   = wave & 3;                 // which 16-row subtile (0..3)
    const int ctile  = wave >> 2;                // which 16-col stripe  (0..3)
    const int mlane  = lane & 15;
    const int khalfA = (lane < 16) ? 0 : 16;     // A-frag K offset per lane half
    const int khB    = (lane < 16) ? 0 : 16;     // B-frag K offset per lane half

    const int slab_base = blockIdx.y * SLAB;
    const int slab_end  = min(slab_base + SLAB, VOCAB);

    float rmax[8], rsum[8];
    #pragma unroll
    for (int r = 0; r < 8; ++r) { rmax[r] = -INFINITY; rsum[r] = 0.0f; }

    const __bf16* arow = xt + (size_t)(rsub * 16 + mlane) * HID;

    for (int c0 = slab_base + ctile * 16; c0 < slab_end; c0 += 64) {
        int col  = c0 + mlane;                       // this lane's vocab column
        int colc = col < VOCAB ? col : (VOCAB - 1);  // clamp for safe loads
        const WT* wrow = w + (size_t)colc * HID;

        // Prefetch next column tile's weight row for this lane.
        {
            int nc = col + 64; nc = nc < VOCAB ? nc : (VOCAB - 1);
            __builtin_prefetch(w + (size_t)nc * HID, 0, 1);
        }

        v8f acc = {};
        #pragma unroll 4
        for (int kb = 0; kb < HID; kb += 32) {
            ABfrag a, b;
            // A: two contiguous 16B LDS loads per lane (ds_load_b128).
            a.h[0] = *(const v8bf*)(arow + kb + khalfA);
            a.h[1] = *(const v8bf*)(arow + kb + 8 + khalfA);
            // B: 16 contiguous K elems of this lane's weight row.
            load_bfrag(b, wrow, kb + khB);
            acc = __builtin_amdgcn_wmma_f32_16x16x32_bf16(
                false, a.v, false, b.v, (short)0, acc, false, false);
        }

        if (col < slab_end) {      // online logsumexp update, 8 rows per lane
            #pragma unroll
            for (int r = 0; r < 8; ++r) {
                float l  = acc[r];
                float nm = fmaxf(rmax[r], l);
                rsum[r]  = rsum[r] * __expf(rmax[r] - nm) + __expf(l - nm);
                rmax[r]  = nm;
            }
        }
    }

    // Reduce across the 16 lanes sharing the same rows (xor stays in-half).
    #pragma unroll
    for (int r = 0; r < 8; ++r) {
        #pragma unroll
        for (int mask = 1; mask < 16; mask <<= 1) {
            float om = __shfl_xor(rmax[r], mask, 32);
            float os = __shfl_xor(rsum[r], mask, 32);
            float nm = fmaxf(rmax[r], om);
            rsum[r]  = rsum[r] * __expf(rmax[r] - nm) + os * __expf(om - nm);
            rmax[r]  = nm;
        }
    }

    // Lane 0 holds rows rsub*16+0..7, lane 16 holds rows rsub*16+8..15.
    if (lane == 0 || lane == 16) {
        int half = lane >> 4;
        #pragma unroll
        for (int r = 0; r < 8; ++r) {
            int row = rsub * 16 + half * 8 + r;
            redmax[wave][row] = rmax[r];
            redsum[wave][row] = rsum[r];
        }
    }
    __syncthreads();

    // Combine the 4 waves covering each row; emit slab partials.
    if (tid < ROWS) {
        int row = tid;
        int r0  = row >> 4;                       // rsub owning this row
        float gm = -INFINITY, gs = 0.0f;
        #pragma unroll
        for (int k = 0; k < WAVES / 4; ++k) {
            int ww = r0 + 4 * k;
            float m = redmax[ww][row], s = redsum[ww][row];
            float nm = fmaxf(gm, m);
            gs = gs * __expf(gm - nm) + s * __expf(m - nm);
            gm = nm;
        }
        size_t idx = (size_t)(nb + row) * NSLAB + blockIdx.y;
        pmax[idx] = gm;
        psum[idx] = gs;
    }
}

// ---------------------------------------------------------------------------
// Kernel 2: combine slab partials + fp32 target-logit dot, write [B, S] loss.
// One wave per token.
// ---------------------------------------------------------------------------
__global__ __launch_bounds__(256) void cce_finalize_kernel(
    const float* __restrict__ x,
    const float* __restrict__ w,
    const int*   __restrict__ labels,   // [BATCH, SEQ]
    const float* __restrict__ pmax,
    const float* __restrict__ psum,
    float* __restrict__ out)            // [BATCH, SEQ]
{
    int lane = threadIdx.x & 31;
    int wv   = threadIdx.x >> 5;
    int n    = blockIdx.x * 8 + wv;
    if (n >= NTOK) return;
    int s = n >> 1, b = n & 1;

    int label = labels[b * SEQ + s];
    int lc = label; lc = lc < 0 ? 0 : lc; lc = lc >= VOCAB ? VOCAB - 1 : lc;

    const float* xr = x + (size_t)n  * HID;
    const float* wr = w + (size_t)lc * HID;
    float dot = 0.0f;
    for (int k = lane * 4; k < HID; k += 32 * 4) {
        float4 xv = *(const float4*)(xr + k);
        float4 wl = *(const float4*)(wr + k);
        dot = fmaf(xv.x, wl.x, dot);
        dot = fmaf(xv.y, wl.y, dot);
        dot = fmaf(xv.z, wl.z, dot);
        dot = fmaf(xv.w, wl.w, dot);
    }
    #pragma unroll
    for (int mask = 1; mask < 32; mask <<= 1) dot += __shfl_xor(dot, mask, 32);

    if (lane == 0) {
        float gm = -INFINITY, gs = 0.0f;
        #pragma unroll
        for (int j = 0; j < NSLAB; ++j) {
            float m = pmax[(size_t)n * NSLAB + j];
            float ss = psum[(size_t)n * NSLAB + j];
            float nm = fmaxf(gm, m);
            gs = gs * __expf(gm - nm) + ss * __expf(m - nm);
            gm = nm;
        }
        float lse = gm + __logf(gs);
        out[b * SEQ + s] = (label == IGNORE_INDEX) ? 0.0f : (lse - dot);
    }
}

extern "C" void kernel_launch(void* const* d_in, const int* in_sizes, int n_in,
                              void* d_out, int out_size, void* d_ws, size_t ws_size,
                              hipStream_t stream) {
    const float* x      = (const float*)d_in[0];   // [SEQ, BATCH, HID] fp32
    const float* w      = (const float*)d_in[1];   // [VOCAB, HID] fp32
    const int*   labels = (const int*)d_in[2];     // [BATCH, SEQ] int32
    float*       out    = (float*)d_out;           // [BATCH, SEQ] fp32

    // Workspace layout: [pmax | psum | (optional) bf16 weight copy]
    float* pmax = (float*)d_ws;                    // [NTOK, NSLAB]
    float* psum = pmax + (size_t)NTOK * NSLAB;     // [NTOK, NSLAB]
    const size_t part_bytes = (size_t)NTOK * NSLAB * 2 * sizeof(float);
    const size_t wbf_bytes  = (size_t)VOCAB * HID * sizeof(__bf16);

    dim3 grid1(NTOK / ROWS, NSLAB);                // 64 x 8 token-tile x slab

    if (ws_size >= part_bytes + wbf_bytes) {
        // Pre-convert W -> bf16: halves HBM stream and all L2-side traffic.
        __bf16* wbf = (__bf16*)((char*)d_ws + part_bytes);
        wcvt_kernel<<<VOCAB, 256, 0, stream>>>(w, wbf);
        cce_partial_kernel<__bf16><<<grid1, TPB, 0, stream>>>(x, wbf, pmax, psum);
    } else {
        // Fallback: stream fp32 weight, convert in registers.
        cce_partial_kernel<float><<<grid1, TPB, 0, stream>>>(x, w, pmax, psum);
    }

    cce_finalize_kernel<<<NTOK / 8, 256, 0, stream>>>(x, w, labels, pmax, psum, out);
}
